// CPCLoss_4363686773177
// MI455X (gfx1250) — matrix operand order repaired and checked by reading, not compiled, not run on hardware
//
#include <hip/hip_runtime.h>
#include <hip/hip_bf16.h>
#include <stdint.h>

typedef __attribute__((ext_vector_type(16))) _Float16     v16h;
typedef __attribute__((ext_vector_type(8)))  _Float16     v8h;
typedef __attribute__((ext_vector_type(8)))  float        v8f;
typedef __attribute__((ext_vector_type(4)))  unsigned int v4u;
typedef __attribute__((ext_vector_type(8)))  int          v8i;
typedef __attribute__((ext_vector_type(4)))  int          v4i;

#define BROWS   8192
#define HDIM    256
#define NROWS   16384          // 2*BROWS
#define INV_TAU 2.0f           // 1/0.5
#define JT      64             // columns per TDM tile (4 x 16-col subtiles)
#define HPAD    264            // padded halves per LDS row (256 + 8 -> 528B stride, bank-conflict free)
#define WAVES   4              // waves per block in main kernel

// ---------------------------------------------------------------------------
// Phase 1: row L2-norms, f16 Z = [x_norm; y_norm], pos[i] = x_n[i].y_n[i].
// One wave32 per row pair; lane handles 8 contiguous floats.
// ---------------------------------------------------------------------------
__global__ __launch_bounds__(256)
void nt_normalize(const float* __restrict__ x, const float* __restrict__ y,
                  _Float16* __restrict__ Z, float* __restrict__ pos,
                  float* __restrict__ out)
{
    if (blockIdx.x == 0 && threadIdx.x == 0) out[0] = 0.0f;   // zero the scalar accumulator
    const int    w    = (blockIdx.x * blockDim.x + threadIdx.x) >> 5;  // row-pair id
    const int    lane = threadIdx.x & 31;
    const size_t base = (size_t)w * HDIM + (size_t)lane * 8;

    float ax[8], ay[8];
    *(float4*)&ax[0] = *(const float4*)(x + base);
    *(float4*)&ax[4] = *(const float4*)(x + base + 4);
    *(float4*)&ay[0] = *(const float4*)(y + base);
    *(float4*)&ay[4] = *(const float4*)(y + base + 4);

    float sx = 0.f, sy = 0.f, sxy = 0.f;
#pragma unroll
    for (int t = 0; t < 8; ++t) {
        sx  += ax[t] * ax[t];
        sy  += ay[t] * ay[t];
        sxy += ax[t] * ay[t];
    }
#pragma unroll
    for (int m = 1; m < 32; m <<= 1) {     // wave32 butterfly reduction
        sx  += __shfl_xor(sx,  m);
        sy  += __shfl_xor(sy,  m);
        sxy += __shfl_xor(sxy, m);
    }
    const float rnx = rsqrtf(sx), rny = rsqrtf(sy);

    v8h hx, hy;
#pragma unroll
    for (int t = 0; t < 8; ++t) {
        hx[t] = (_Float16)(ax[t] * rnx);
        hy[t] = (_Float16)(ay[t] * rny);
    }
    *(v8h*)(Z + base)                        = hx;   // row w
    *(v8h*)(Z + (size_t)BROWS * HDIM + base) = hy;   // row w + B
    if (lane == 0) pos[w] = sxy * rnx * rny;
}

// ---------------------------------------------------------------------------
// TDM: DMA a JT x 256 f16 tile of Z into LDS, padding 16B per 512B row
// (pad_interval=128 DWORDs, pad_amount=4 DWORDs) -> 528B LDS row stride.
// D# layout per CDNA5 ISA 8.3/8.4. 6-arg builtin (clang-23 / therock headers).
// ---------------------------------------------------------------------------
__device__ __forceinline__ void tdm_issue(const _Float16* gsrc, unsigned ldsAddr)
{
    const unsigned long long ga = (unsigned long long)(uintptr_t)gsrc;
    v4u g0;
    g0[0] = 1u;                                   // count=1, user descriptor, no gather
    g0[1] = ldsAddr;                              // lds_addr (bytes)
    g0[2] = (unsigned)ga;                         // global_addr[31:0]
    g0[3] = (unsigned)(ga >> 32) | (2u << 30);    // global_addr[56:32] | type=2 (image)
    v8i g1;
    g1[0] = (int)((1u << 16)                      // data_size = 1 -> 2 bytes
                | (1u << 20)                      // pad_enable
                | (6u << 22)                      // pad_interval code 6 = 128 DWORDs (one 512B row)
                | (3u << 25));                    // pad_amount  code 3 = 4 DWORDs (16B)
    g1[1] = (int)(((unsigned)HDIM  & 0xFFFFu) << 16);  // tensor_dim0 lo16 (256)
    g1[2] = (int)(((unsigned)NROWS & 0xFFFFu) << 16);  // tensor_dim1 lo16 (16384)
    g1[3] = (int)(((unsigned)HDIM) << 16);             // tile_dim0 = 256
    g1[4] = JT;                                        // tile_dim1 = 64 rows
    g1[5] = HDIM;                                      // tensor_dim0_stride = 256
    g1[6] = 0;
    g1[7] = 0;
    const v4i gz4 = {0, 0, 0, 0};                      // 2D tensor: groups 2/3 unused
    const v8i gz8 = {0, 0, 0, 0, 0, 0, 0, 0};
    __builtin_amdgcn_tensor_load_to_lds(g0, g1, gz4, gz4, gz8, 0);
}

// ---------------------------------------------------------------------------
// Phase 2: fused Z@Z^T -> exp -> (rowsum - diag) -> log -> mean.
// Each wave owns 16 rows (A panel resident in 8 v16h fragments, K=256).
// Block streams all 16384 columns through double-buffered TDM LDS tiles.
// Inner loop: two 16-col subtiles at once with software-pipelined B loads.
// ---------------------------------------------------------------------------
__global__ __launch_bounds__(WAVES * 32)
void nt_main(const _Float16* __restrict__ Z, const float* __restrict__ pos,
             float* __restrict__ out)
{
    __shared__ _Float16 smem[2][JT * HPAD];

    const int lane = threadIdx.x & 31;
    const int wave = threadIdx.x >> 5;
    const int hi   = lane >> 4;           // half-wave select
    const int n    = lane & 15;           // column-within-subtile / row-within-panel
    const int i0   = (blockIdx.x * WAVES + wave) * 16;

    // Resident A fragments: lane holds 16 contiguous K of row n (lanes>=16: K+16)
    v16h afrag[8];
    {
        const _Float16* rowp = Z + (size_t)(i0 + n) * HDIM;
#pragma unroll
        for (int kc = 0; kc < 8; ++kc) {
            const int k0 = kc * 32 + hi * 16;
            v8h lo = *(const v8h*)(rowp + k0);
            v8h hh = *(const v8h*)(rowp + k0 + 8);
            v16h a;
#pragma unroll
            for (int t = 0; t < 8; ++t) { a[t] = lo[t]; a[t + 8] = hh[t]; }
            afrag[kc] = a;
        }
    }

    float rowAcc[8] = {0.f, 0.f, 0.f, 0.f, 0.f, 0.f, 0.f, 0.f};

    if (threadIdx.x == 0)                          // TDM issues regardless of EXEC;
        tdm_issue(Z, (unsigned)(uintptr_t)&smem[0][0]);  // scalar branch keeps it to wave 0

    const int ntiles = NROWS / JT;                 // 256
    for (int t = 0; t < ntiles; ++t) {
        if (threadIdx.x == 0) __builtin_amdgcn_s_wait_tensorcnt(0);
        __syncthreads();                           // tile t resident in smem[t&1]
        if (threadIdx.x == 0 && (t + 1) < ntiles)  // prefetch next tile into other buffer
            tdm_issue(Z + (size_t)(t + 1) * JT * HDIM,
                      (unsigned)(uintptr_t)&smem[(t + 1) & 1][0]);

        const _Float16* bt = &smem[t & 1][0];
#pragma unroll
        for (int sp = 0; sp < JT / 32; ++sp) {     // process subtiles in pairs
            // B fragments: lane holds 16 contiguous K of column n (lanes>=16: K+16)
            const _Float16* colp0 = bt + (size_t)(sp * 32 + n)      * HPAD + hi * 16;
            const _Float16* colp1 = bt + (size_t)(sp * 32 + 16 + n) * HPAD + hi * 16;
            v8f c0 = {0.f, 0.f, 0.f, 0.f, 0.f, 0.f, 0.f, 0.f};
            v8f c1 = {0.f, 0.f, 0.f, 0.f, 0.f, 0.f, 0.f, 0.f};
            v16h b0 = *(const v16h*)(colp0);       // software-pipelined B stream
            v16h b1 = *(const v16h*)(colp1);
#pragma unroll
            for (int kc = 0; kc < 8; ++kc) {
                v16h nb0, nb1;
                if (kc < 7) {                      // issue next loads before this WMMA pair
                    nb0 = *(const v16h*)(colp0 + (kc + 1) * 32);
                    nb1 = *(const v16h*)(colp1 + (kc + 1) * 32);
                }
                c0 = __builtin_amdgcn_wmma_f32_16x16x32_f16(
                         false, afrag[kc], false, b0, (short)0, c0, false, false);
                c1 = __builtin_amdgcn_wmma_f32_16x16x32_f16(
                         false, afrag[kc], false, b1, (short)0, c1, false, false);
                b0 = nb0;
                b1 = nb1;
            }
            // c[v]: row i0 + v + 8*hi, column jcol -> exp, drop diagonal, sum columns
            const int jcol0 = t * JT + sp * 32 + n;
#pragma unroll
            for (int v = 0; v < 8; ++v) {
                const int row = i0 + v + hi * 8;
                float e0 = __expf(c0[v] * INV_TAU);
                float e1 = __expf(c1[v] * INV_TAU);
                if (jcol0      == row) e0 = 0.f;   // rowsum - diag, exactly
                if (jcol0 + 16 == row) e1 = 0.f;
                float e = e0 + e1;
                e += __shfl_xor(e, 1);
                e += __shfl_xor(e, 2);
                e += __shfl_xor(e, 4);
                e += __shfl_xor(e, 8);             // sum within each 16-lane column group
                rowAcc[v] += e;
            }
        }
    }

    // per-row: log(denom) - pos/tau ; reduce the wave's 16 rows, add to mean
    float s = 0.f;
    if (n == 0) {                                  // one lane per 16-group (values replicated)
#pragma unroll
        for (int v = 0; v < 8; ++v) {
            const int row = i0 + v + hi * 8;
            s += __logf(rowAcc[v]) - INV_TAU * pos[row & (BROWS - 1)];
        }
    }
    s += __shfl_xor(s, 16);                        // combine the two half-wave row groups
    if (lane == 0) atomicAdd(out, s * (1.0f / (float)NROWS));
}

// ---------------------------------------------------------------------------
extern "C" void kernel_launch(void* const* d_in, const int* in_sizes, int n_in,
                              void* d_out, int out_size, void* d_ws, size_t ws_size,
                              hipStream_t stream)
{
    (void)in_sizes; (void)n_in; (void)out_size; (void)ws_size;
    const float* x = (const float*)d_in[0];
    const float* y = (const float*)d_in[1];
    float* out = (float*)d_out;

    // workspace layout: Z (16384*256 f16 = 8 MB) | pos (8192 f32 = 32 KB)
    _Float16* Z   = (_Float16*)d_ws;
    float*    pos = (float*)((char*)d_ws + (size_t)NROWS * HDIM * sizeof(_Float16));

    nt_normalize<<<BROWS / 8, 256, 0, stream>>>(x, y, Z, pos, out);
    nt_main<<<NROWS / (WAVES * 16), WAVES * 32, 0, stream>>>(Z, pos, out);
}
